// QuantMLP_14637248545468
// MI455X (gfx1250) — compile-verified, matrix operand order checked by reading, not run on hardware
//
#include <hip/hip_runtime.h>
#include <hip/hip_bf16.h>
#include <math.h>

typedef __attribute__((ext_vector_type(8))) int v8i;
typedef int vec4i __attribute__((vector_size(16)));
typedef __attribute__((address_space(1))) vec4i gvec4i;   // global (__device__) int4
typedef __attribute__((address_space(3))) vec4i lvec4i;   // LDS (__shared__) int4

#define BM 128
#define BN 256
#define BK 64
#define ROWB 80          // LDS row stride in bytes: 64 data + 16 pad (16B aligned)
#define THREADS 256

// ---------------------------------------------------------------------------
// CDNA5 async global->LDS copy (GLOBAL_LOAD_ASYNC_TO_LDS_B128, ASYNCcnt)
// ---------------------------------------------------------------------------
__device__ __forceinline__ void async_copy16(const void* g, void* l) {
#if __has_builtin(__builtin_amdgcn_global_load_async_to_lds_b128)
    __builtin_amdgcn_global_load_async_to_lds_b128(
        (gvec4i*)g, (lvec4i*)l, 0, 0);
#else
    unsigned lo = (unsigned)(unsigned long long)
        (__attribute__((address_space(3))) char*)l;   // LDS byte offset
    asm volatile("global_load_async_to_lds_b128 %0, %1, off"
                 :: "v"(lo), "v"(g) : "memory");
#endif
}

__device__ __forceinline__ void wait_async0() {
#if __has_builtin(__builtin_amdgcn_s_wait_asynccnt)
    __builtin_amdgcn_s_wait_asynccnt(0);
#else
    asm volatile("s_wait_asynccnt 0x0" ::: "memory");
#endif
}

// ---------------------------------------------------------------------------
// int32 -> int8 pack (values already in [-127,127]; low byte == int8 repr)
// ---------------------------------------------------------------------------
__global__ __launch_bounds__(THREADS)
void pack_i8_kernel(const int* __restrict__ src, unsigned int* __restrict__ dst, int n4) {
    int i = blockIdx.x * blockDim.x + threadIdx.x;
    if (i < n4) {
        const int4 v = reinterpret_cast<const int4*>(src)[i];
        unsigned int b = (unsigned int)(v.x & 0xff)
                       | ((unsigned int)(v.y & 0xff) << 8)
                       | ((unsigned int)(v.z & 0xff) << 16)
                       | ((unsigned int)(v.w & 0xff) << 24);
        dst[i] = b;
    }
}

// ---------------------------------------------------------------------------
// Load one 8-bit A/B WMMA fragment (16x64) from LDS.
// Per ISA 7.12.2: lanes 0-15 hold K={0-7,16-23,32-39,48-55} of row (lane&15),
// lanes 16-31 hold K shifted by +8. 4x 8-byte chunks at +0,+16,+32,+48.
// ---------------------------------------------------------------------------
__device__ __forceinline__ v8i load_frag(const signed char* base) {
    const int2 c0 = *reinterpret_cast<const int2*>(base +  0);
    const int2 c1 = *reinterpret_cast<const int2*>(base + 16);
    const int2 c2 = *reinterpret_cast<const int2*>(base + 32);
    const int2 c3 = *reinterpret_cast<const int2*>(base + 48);
    v8i f;
    f[0] = c0.x; f[1] = c0.y;
    f[2] = c1.x; f[3] = c1.y;
    f[4] = c2.x; f[5] = c2.y;
    f[6] = c3.x; f[7] = c3.y;
    return f;
}

// ---------------------------------------------------------------------------
// Tiled int8 GEMM: C[row, col] = sum_k A[row,k] * B[col,k]   (both K-major)
// Block tile 128x256, 8 waves, each wave computes a 64x64 tile (16 WMMAs/step).
// MODE 0: y = acc*rowScale[row]*colScale[col] + bias[col]; out = gelu_exact(y)
// MODE 1: y = acc*rowScale[row]*colScale[col] + bias[col];  out = y
// ---------------------------------------------------------------------------
template <int MODE>
__global__ __launch_bounds__(THREADS)
void gemm_iu8_kernel(const signed char* __restrict__ A,
                     const signed char* __restrict__ B,
                     int K, int NCOLS,
                     const float* __restrict__ rowScale,
                     const float* __restrict__ colScale,
                     const float* __restrict__ bias,
                     float* __restrict__ Out)
{
    __shared__ signed char sA[2][BM * ROWB];   // 20 KiB
    __shared__ signed char sB[2][BN * ROWB];   // 40 KiB

    const int tid   = threadIdx.x;
    const int lane  = tid & 31;
    const int wid   = tid >> 5;
    const int half  = lane >> 4;     // 0: lanes 0-15, 1: lanes 16-31
    const int r     = lane & 15;
    const int waveM = wid & 1;       // 2 waves down   -> 64-row strips
    const int waveN = wid >> 1;      // 4 waves across -> 64-col strips

    const int rowBase = blockIdx.y * BM;
    const int colBase = blockIdx.x * BN;

    // Async staging assignment:
    //   A tile: 128 rows x 64B -> 2 threads/row, 32B each (2x b128)
    //   B tile: 256 rows x 64B -> 1 thread/row, 64B each (4x b128)
    const int ldRowA = tid >> 1;
    const int ldOffA = (tid & 1) * 32;
    const signed char* gA = A + (size_t)(rowBase + ldRowA) * K + ldOffA;
    const signed char* gB = B + (size_t)(colBase + tid) * K;

    const int nSteps = K / BK;

    // Issue tile 0 into buffer 0.
    {
        signed char* dA = &sA[0][ldRowA * ROWB + ldOffA];
        async_copy16(gA,      dA);
        async_copy16(gA + 16, dA + 16);
        signed char* dB = &sB[0][tid * ROWB];
        async_copy16(gB,      dB);
        async_copy16(gB + 16, dB + 16);
        async_copy16(gB + 32, dB + 32);
        async_copy16(gB + 48, dB + 48);
    }

    v8i acc[4][4] = {};

    for (int s = 0; s < nSteps; ++s) {
        const int cur = s & 1;

        wait_async0();        // tile s resident in LDS (this wave's portion)
        __syncthreads();      // all waves done: tile s visible, buf cur^1 free

        if (s + 1 < nSteps) { // overlap tile s+1 DMA with tile s compute
            const size_t ko = (size_t)(s + 1) * BK;
            signed char* dA = &sA[cur ^ 1][ldRowA * ROWB + ldOffA];
            async_copy16(gA + ko,      dA);
            async_copy16(gA + ko + 16, dA + 16);
            signed char* dB = &sB[cur ^ 1][tid * ROWB];
            async_copy16(gB + ko,      dB);
            async_copy16(gB + ko + 16, dB + 16);
            async_copy16(gB + ko + 32, dB + 32);
            async_copy16(gB + ko + 48, dB + 48);
            if (s + 2 < nSteps) {
                __builtin_prefetch(gA + (size_t)(s + 2) * BK, 0, 1);
                __builtin_prefetch(gB + (size_t)(s + 2) * BK, 0, 1);
            }
        }

        v8i af[4], bf[4];
        #pragma unroll
        for (int t = 0; t < 4; ++t)
            af[t] = load_frag(&sA[cur][(waveM * 64 + t * 16 + r) * ROWB + half * 8]);
        #pragma unroll
        for (int u = 0; u < 4; ++u)
            bf[u] = load_frag(&sB[cur][(waveN * 64 + u * 16 + r) * ROWB + half * 8]);

        #pragma unroll
        for (int t = 0; t < 4; ++t)
            #pragma unroll
            for (int u = 0; u < 4; ++u)
                acc[t][u] = __builtin_amdgcn_wmma_i32_16x16x64_iu8(
                    true, af[t], true, bf[u], acc[t][u], false, false);
    }

    // Epilogue. i32 16x16 C layout: VGPR v, lane<16 -> M=v, lane>=16 -> M=v+8.
    #pragma unroll
    for (int t = 0; t < 4; ++t) {
        const int gRow0 = rowBase + waveM * 64 + t * 16 + half * 8;
        float rs[8];
        #pragma unroll
        for (int vr = 0; vr < 8; ++vr) rs[vr] = rowScale[gRow0 + vr];
        #pragma unroll
        for (int u = 0; u < 4; ++u) {
            const int gCol = colBase + waveN * 64 + u * 16 + r;
            const float cs = colScale[gCol];
            const float bv = bias[gCol];
            #pragma unroll
            for (int vr = 0; vr < 8; ++vr) {
                float y = (float)acc[t][u][vr] * rs[vr] * cs + bv;
                if (MODE == 0) {
                    y = 0.5f * y * (1.0f + erff(y * 0.70710678118654752f)); // exact GELU
                }
                Out[(size_t)(gRow0 + vr) * NCOLS + gCol] = y;
            }
        }
    }
}

// ---------------------------------------------------------------------------
// Per-row dynamic quantization: s = max(|row|)/127 (>=1e-8); q = clip(rint(x/s))
// ---------------------------------------------------------------------------
__global__ __launch_bounds__(THREADS)
void rowquant_kernel(const float* __restrict__ act,
                     signed char* __restrict__ q,
                     float* __restrict__ s2, int F)
{
    __shared__ float red[THREADS];
    const int row = blockIdx.x;
    const float* a = act + (size_t)row * F;

    float m = 0.0f;
    for (int i = threadIdx.x * 4; i < F; i += THREADS * 4) {
        const float4 v = *reinterpret_cast<const float4*>(a + i);
        m = fmaxf(m, fmaxf(fmaxf(fabsf(v.x), fabsf(v.y)),
                           fmaxf(fabsf(v.z), fabsf(v.w))));
    }
    red[threadIdx.x] = m;
    __syncthreads();
    for (int off = THREADS / 2; off > 0; off >>= 1) {
        if (threadIdx.x < off)
            red[threadIdx.x] = fmaxf(red[threadIdx.x], red[threadIdx.x + off]);
        __syncthreads();
    }
    const float s   = fmaxf(red[0] * (1.0f / 127.0f), 1e-8f);
    const float inv = 1.0f / s;
    if (threadIdx.x == 0) s2[row] = s;

    unsigned int* qr = reinterpret_cast<unsigned int*>(q + (size_t)row * F);
    for (int i = threadIdx.x * 4; i < F; i += THREADS * 4) {
        const float4 v = *reinterpret_cast<const float4*>(a + i);
        int x0 = (int)rintf(v.x * inv);
        int x1 = (int)rintf(v.y * inv);
        int x2 = (int)rintf(v.z * inv);
        int x3 = (int)rintf(v.w * inv);
        x0 = min(127, max(-128, x0));
        x1 = min(127, max(-128, x1));
        x2 = min(127, max(-128, x2));
        x3 = min(127, max(-128, x3));
        qr[i >> 2] = (unsigned int)(x0 & 0xff)
                   | ((unsigned int)(x1 & 0xff) << 8)
                   | ((unsigned int)(x2 & 0xff) << 16)
                   | ((unsigned int)(x3 & 0xff) << 24);
    }
}

// ---------------------------------------------------------------------------
extern "C" void kernel_launch(void* const* d_in, const int* in_sizes, int n_in,
                              void* d_out, int out_size, void* d_ws, size_t ws_size,
                              hipStream_t stream)
{
    (void)in_sizes; (void)n_in; (void)out_size; (void)ws_size;
    const int N = 4096, D = 4096, F = 16384;

    const int*   q_in     = (const int*)  d_in[0];
    const float* in_scale = (const float*)d_in[1];
    const int*   w1_q     = (const int*)  d_in[2];
    const float* w1_scale = (const float*)d_in[3];
    const float* b1       = (const float*)d_in[4];
    const int*   w2_q     = (const int*)  d_in[5];
    const float* w2_scale = (const float*)d_in[6];
    const float* b2       = (const float*)d_in[7];
    float*       out      = (float*)d_out;

    char* ws = (char*)d_ws;
    const size_t SZ_A8 = (size_t)N * D;   // 16 MiB
    const size_t SZ_W1 = (size_t)F * D;   // 64 MiB
    const size_t SZ_W2 = (size_t)D * F;   // 64 MiB
    signed char* A8   = (signed char*)(ws);
    signed char* W1   = (signed char*)(ws + SZ_A8);
    signed char* W2   = (signed char*)(ws + SZ_A8 + SZ_W1);
    float*       actF = (float*)      (ws + SZ_A8 + SZ_W1 + SZ_W2);  // N*F fp32 (256 MiB)
    // Stream-ordered reuse: fc1 is done with W1 / A8 before these are written.
    signed char* Q2   = W1;               // N*F int8 == F*D bytes
    float*       S2   = (float*)A8;       // N fp32

    const dim3 blk(THREADS);

    int n4 = (N * D) / 4;
    pack_i8_kernel<<<(n4 + THREADS - 1) / THREADS, blk, 0, stream>>>(q_in, (unsigned int*)A8, n4);
    n4 = (F * D) / 4;
    pack_i8_kernel<<<(n4 + THREADS - 1) / THREADS, blk, 0, stream>>>(w1_q, (unsigned int*)W1, n4);
    n4 = (D * F) / 4;
    pack_i8_kernel<<<(n4 + THREADS - 1) / THREADS, blk, 0, stream>>>(w2_q, (unsigned int*)W2, n4);

    // fc1: (N x D) * (F x D)^T -> act (N x F), fused dequant+bias+GELU
    const dim3 g1(F / BN, N / BM);
    gemm_iu8_kernel<0><<<g1, blk, 0, stream>>>(A8, W1, D, F, in_scale, w1_scale, b1, actF);

    // dynamic per-token requant
    rowquant_kernel<<<N, blk, 0, stream>>>(actF, Q2, S2, F);

    // fc2: (N x F) * (D x F)^T -> out (N x D), fused dequant+bias
    const dim3 g2(D / BN, N / BM);
    gemm_iu8_kernel<1><<<g2, blk, 0, stream>>>(Q2, W2, F, D, S2, w2_scale, b2, out);
}
